// TinyPI05V2Pytorch_79766132621536
// MI455X (gfx1250) — matrix operand order repaired and checked by reading, not compiled, not run on hardware
//
#include <hip/hip_runtime.h>

// ---------------------------------------------------------------------------
// MI455X (gfx1250) implementation: bf16 WMMA GEMM pipeline for the fused
// dual-stream transformer block.
//   * All heavy GEMMs -> v_wmma_f32_16x16x32_bf16 (fp32 accumulate).
//   * Global->LDS via GLOBAL_LOAD_ASYNC_TO_LDS_B128 (ASYNCcnt) on every full
//     K-tile; sync vmem + ds_store fallback only for the K-tail tile.
//   * V is pre-transposed once (LDS-tiled) so attn@V also uses the fast path.
//   * 128x128x64 block tile, 8 waves (4Mx2N), each wave 32x64 = 8 WMMA accs.
//   * Double-buffered dynamic LDS (73.7KB of the WGP's 320KB -> 4 blocks/WGP).
// Roofline: ~1.5 TFLOP of GEMM vs ~260MB bf16 weights (fits 192MB L2);
// matrix-pipe bound, so everything routes through WMMA.
// ---------------------------------------------------------------------------

typedef __bf16  bf16x8  __attribute__((ext_vector_type(8)));
typedef __bf16  v16bf   __attribute__((ext_vector_type(16)));
typedef float   v8f     __attribute__((ext_vector_type(8)));

// ------------------------------- constants ---------------------------------
#define C_B   8
#define C_SP  800
#define C_SS  48
#define C_N   848       // SP + SS = 53*16
#define C_DP  2048
#define C_DS  1024
#define C_DC  1024
#define C_H   8
#define C_HD  256
#define C_HH  2048      // H*HD
#define C_FP  16384
#define C_FS  4096

// --------------------------- elementwise kernels ---------------------------

__global__ void k_f32_to_bf16(const float* __restrict__ in, __bf16* __restrict__ out, long n) {
    long i  = (long)blockIdx.x * blockDim.x + threadIdx.x;
    long st = (long)gridDim.x * blockDim.x;
    for (; i < n; i += st) out[i] = (__bf16)in[i];
}

// Tiled bf16 transpose: in [z][R][C] -> out [z][C][R]
__global__ void k_transpose_bf16(const __bf16* __restrict__ in, __bf16* __restrict__ out,
                                 int R, int Cc, long sIn, long sOut) {
    __shared__ __bf16 t[32][33];
    int r0 = blockIdx.y * 32, c0 = blockIdx.x * 32;
    const __bf16* ib = in  + (long)blockIdx.z * sIn;
    __bf16*       ob = out + (long)blockIdx.z * sOut;
    int tx = threadIdx.x & 31, ty = threadIdx.x >> 5;   // 8 rows per pass
    for (int i = ty; i < 32; i += 8) {
        int r = r0 + i, c = c0 + tx;
        t[i][tx] = (r < R && c < Cc) ? ib[(long)r * Cc + c] : (__bf16)0.f;
    }
    __syncthreads();
    for (int i = ty; i < 32; i += 8) {
        int c = c0 + i, r = r0 + tx;
        if (c < Cc && r < R) ob[(long)c * R + r] = t[tx][i];
    }
}

// RMSNorm * (1 + w)  ->  bf16.  One block per row.
__global__ void k_gemma_norm(const float* __restrict__ x, const float* __restrict__ w,
                             __bf16* __restrict__ out, int D) {
    int row = blockIdx.x;
    const float* xr = x + (long)row * D;
    __shared__ float red[256];
    float s = 0.f;
    for (int c = threadIdx.x; c < D; c += 256) { float v = xr[c]; s += v * v; }
    red[threadIdx.x] = s; __syncthreads();
    for (int o = 128; o; o >>= 1) {
        if (threadIdx.x < o) red[threadIdx.x] += red[threadIdx.x + o];
        __syncthreads();
    }
    float inv = rsqrtf(red[0] / (float)D + 1e-6f);
    for (int c = threadIdx.x; c < D; c += 256)
        out[(long)row * D + c] = (__bf16)(xr[c] * inv * (1.f + w[c]));
}

// mod = cond @ w.T + b   (tiny: 8 x 3072, K=1024)
__global__ void k_mod(const float* __restrict__ cond, const float* __restrict__ w,
                      const float* __restrict__ bias, float* __restrict__ out,
                      int Bn, int Nn, int Kn) {
    int i = blockIdx.x * blockDim.x + threadIdx.x;
    if (i >= Bn * Nn) return;
    int b = i / Nn, n = i % Nn;
    const float* c  = cond + (long)b * Kn;
    const float* wr = w    + (long)n * Kn;
    float s = bias[n];
    for (int k = 0; k < Kn; k++) s += c[k] * wr[k];
    out[i] = s;
}

// AdaNorm: out = rms(x)*(1+scale) + shift (bf16); also extracts gate (fp32).
__global__ void k_ada_norm(const float* __restrict__ x, const float* __restrict__ mod,
                           __bf16* __restrict__ out, float* __restrict__ gate,
                           int S, int D) {
    int row = blockIdx.x;        // B*S rows
    int b   = row / S;
    const float* xr = x + (long)row * D;
    __shared__ float red[256];
    float s = 0.f;
    for (int c = threadIdx.x; c < D; c += 256) { float v = xr[c]; s += v * v; }
    red[threadIdx.x] = s; __syncthreads();
    for (int o = 128; o; o >>= 1) {
        if (threadIdx.x < o) red[threadIdx.x] += red[threadIdx.x + o];
        __syncthreads();
    }
    float inv = rsqrtf(red[0] / (float)D + 1e-6f);
    const float* m = mod + (long)b * 3 * D;
    for (int c = threadIdx.x; c < D; c += 256) {
        float v = xr[c] * inv;
        out[(long)row * D + c] = (__bf16)(v * (1.f + m[c]) + m[D + c]);
    }
    if (row % S == 0)
        for (int c = threadIdx.x; c < D; c += 256)
            gate[(long)b * D + c] = m[2 * D + c];
}

// RoPE in place on bf16 (layout [B, S, Hn, hd]); position_ids == arange(S).
__global__ void k_rope(__bf16* __restrict__ x, int Bn, int S, int Hn, int hd) {
    int hf = hd / 2;
    long total = (long)Bn * S * Hn * hf;
    long i  = (long)blockIdx.x * blockDim.x + threadIdx.x;
    long st = (long)gridDim.x * blockDim.x;
    for (; i < total; i += st) {
        int  d = (int)(i % hf);
        long t = i / hf;
        int  h = (int)(t % Hn); t /= Hn;
        int  s = (int)(t % S);
        int  b = (int)(t / S);
        float ang = (float)s * __powf(10000.f, -(float)d / (float)hf);
        float sn, cs; __sincosf(ang, &sn, &cs);
        long base = (((long)b * S + s) * Hn + h) * hd;
        float x1 = (float)x[base + d];
        float x2 = (float)x[base + hf + d];
        x[base + d]      = (__bf16)(x1 * cs - x2 * sn);
        x[base + hf + d] = (__bf16)(x2 * cs + x1 * sn);
    }
}

// Masked row softmax: scores fp32 -> attn bf16. Row id = (b*H + h)*N + q.
// Prefix queries (q < SP) only see keys [0, SP); suffix queries see all.
__global__ void k_softmax(const float* __restrict__ scores, __bf16* __restrict__ attn,
                          int Nk, int SPq) {
    long row = blockIdx.x;
    int  q   = (int)(row % Nk);
    int  lim = (q < SPq) ? SPq : Nk;
    const float* sr = scores + row * Nk;
    __bf16*      ar = attn   + row * Nk;
    __shared__ float red[256];
    float mx = -3.0e38f;
    for (int c = threadIdx.x; c < lim; c += 256) mx = fmaxf(mx, sr[c]);
    red[threadIdx.x] = mx; __syncthreads();
    for (int o = 128; o; o >>= 1) {
        if (threadIdx.x < o) red[threadIdx.x] = fmaxf(red[threadIdx.x], red[threadIdx.x + o]);
        __syncthreads();
    }
    mx = red[0]; __syncthreads();
    float sm = 0.f;
    for (int c = threadIdx.x; c < lim; c += 256) sm += __expf(sr[c] - mx);
    red[threadIdx.x] = sm; __syncthreads();
    for (int o = 128; o; o >>= 1) {
        if (threadIdx.x < o) red[threadIdx.x] += red[threadIdx.x + o];
        __syncthreads();
    }
    float inv = 1.f / red[0];
    for (int c = threadIdx.x; c < Nk; c += 256)
        ar[c] = (c < lim) ? (__bf16)(__expf(sr[c] - mx) * inv) : (__bf16)0.f;
}

// hid = gelu_tanh(gate) * up  (bf16 in/out)
__global__ void k_gelumul(const __bf16* __restrict__ g, const __bf16* __restrict__ u,
                          __bf16* __restrict__ out, long n) {
    long i  = (long)blockIdx.x * blockDim.x + threadIdx.x;
    long st = (long)gridDim.x * blockDim.x;
    for (; i < n; i += st) {
        float x  = (float)g[i];
        float t  = 0.7978845608028654f * (x + 0.044715f * x * x * x);
        float ge = 0.5f * x * (1.f + tanhf(t));
        out[i] = (__bf16)(ge * (float)u[i]);
    }
}

// out = a + t
__global__ void k_add(const float* __restrict__ a, const float* __restrict__ t,
                      float* __restrict__ out, long n) {
    long i  = (long)blockIdx.x * blockDim.x + threadIdx.x;
    long st = (long)gridDim.x * blockDim.x;
    for (; i < n; i += st) out[i] = a[i] + t[i];
}

// out = a + t * gate[b, c]   (layout [B, S, D])
__global__ void k_add_gate(const float* __restrict__ a, const float* __restrict__ t,
                           const float* __restrict__ gate, float* __restrict__ out,
                           int S, int D, long n) {
    long i  = (long)blockIdx.x * blockDim.x + threadIdx.x;
    long st = (long)gridDim.x * blockDim.x;
    for (; i < n; i += st) {
        int b = (int)(i / ((long)S * D));
        int c = (int)(i % D);
        out[i] = a[i] + t[i] * gate[(long)b * D + c];
    }
}

// ------------------------------ WMMA GEMM ----------------------------------
// C[M,N] = alpha * A[M,K] @ B^T,  A bf16 row-major (lda),
// B bf16 weight-form (N,K) row-major (ldb = K-row stride).
// Batched via blockIdx.z: off = (z/zdiv)*s1 + (z%zdiv)*s2 (element strides).
// Block tile 128x128x64, 256 threads = 8 waves (4M x 2N); wave tile 32x64
// (2x4 WMMA accumulators). Double-buffered dynamic LDS; global->LDS via
// async-to-LDS on full K-tiles (ASYNCcnt), sync ds_store fallback on tails.
#define BM 128
#define BN 128
#define BK 64
#define LDS_STRIDE 72          // BK + 8 bf16 pad (16B) for bank spread
#define GEMM_LDS_BYTES ((size_t)(2 * BM * LDS_STRIDE + 2 * BN * LDS_STRIDE) * 2)

template <bool OBF>
__global__ __launch_bounds__(256) void k_gemm(
    const __bf16* __restrict__ A, const __bf16* __restrict__ Bm, void* __restrict__ Cv,
    int M, int N, int K, int lda, int ldb, int ldc, float alpha, int zdiv,
    long sA1, long sA2, long sB1, long sB2, long sC1, long sC2) {
    extern __shared__ char dynsmem[];
    __bf16* AsB = (__bf16*)dynsmem;                        // [2][BM][LDS_STRIDE]
    __bf16* WsB = AsB + 2 * BM * LDS_STRIDE;               // [2][BN][LDS_STRIDE]
#define AS(buf, r, c) AsB[((buf) * BM + (r)) * LDS_STRIDE + (c)]
#define WS(buf, r, c) WsB[((buf) * BN + (r)) * LDS_STRIDE + (c)]

    int z = blockIdx.z;
    long offA = (long)(z / zdiv) * sA1 + (long)(z % zdiv) * sA2;
    long offB = (long)(z / zdiv) * sB1 + (long)(z % zdiv) * sB2;
    long offC = (long)(z / zdiv) * sC1 + (long)(z % zdiv) * sC2;
    const __bf16* Ag = A  + offA;
    const __bf16* Bg = Bm + offB;

    int m0 = blockIdx.y * BM, n0 = blockIdx.x * BN;
    int tid = threadIdx.x;

    // --- global -> LDS movers. Rows/cols are clamped (out-of-range output
    // rows/cols are masked at the C store); only K-overrun must be zeroed.
    auto asyncA = [&](int kb, int buf) {
#pragma unroll
        for (int i = 0; i < 4; i++) {
            int c = tid + i * 256;              // 1024 chunks of 8 bf16
            int r = c >> 3, col = (c & 7) * 8;
            int gr = m0 + r; gr = gr < M - 1 ? gr : M - 1;
            const __bf16* gp = Ag + (long)gr * lda + (kb + col);
            unsigned lo = (unsigned)(size_t)&AS(buf, r, col);
            asm volatile("global_load_async_to_lds_b128 %0, %1, off"
                         :: "v"(lo), "v"(gp) : "memory");
        }
    };
    auto syncA = [&](int kb, int buf) {
#pragma unroll
        for (int i = 0; i < 4; i++) {
            int c = tid + i * 256;
            int r = c >> 3, col = (c & 7) * 8;
            int gr = m0 + r; gr = gr < M - 1 ? gr : M - 1;
            int gc = kb + col;
            __bf16* dst = &AS(buf, r, col);
            if (gc + 8 <= K) {
                *(uint4*)dst = *(const uint4*)(Ag + (long)gr * lda + gc);
            } else {
#pragma unroll
                for (int j = 0; j < 8; j++) {
                    int kx = gc + j;
                    __bf16 v = Ag[(long)gr * lda + (kx < K ? kx : 0)];
                    dst[j] = (kx < K) ? v : (__bf16)0.f;
                }
            }
        }
    };
    auto asyncW = [&](int kb, int buf) {
#pragma unroll
        for (int i = 0; i < 4; i++) {
            int c = tid + i * 256;
            int r = c >> 3, col = (c & 7) * 8;
            int gn = n0 + r; gn = gn < N - 1 ? gn : N - 1;
            const __bf16* gp = Bg + (long)gn * ldb + (kb + col);
            unsigned lo = (unsigned)(size_t)&WS(buf, r, col);
            asm volatile("global_load_async_to_lds_b128 %0, %1, off"
                         :: "v"(lo), "v"(gp) : "memory");
        }
    };
    auto syncW = [&](int kb, int buf) {
#pragma unroll
        for (int i = 0; i < 4; i++) {
            int c = tid + i * 256;
            int r = c >> 3, col = (c & 7) * 8;
            int gn = n0 + r; gn = gn < N - 1 ? gn : N - 1;
            int gc = kb + col;
            __bf16* dst = &WS(buf, r, col);
            if (gc + 8 <= K) {
                *(uint4*)dst = *(const uint4*)(Bg + (long)gn * ldb + gc);
            } else {
#pragma unroll
                for (int j = 0; j < 8; j++) {
                    int kx = gc + j;
                    __bf16 v = Bg[(long)gn * ldb + (kx < K ? kx : 0)];
                    dst[j] = (kx < K) ? v : (__bf16)0.f;
                }
            }
        }
    };
    auto issueTile = [&](int t2, int buf) {
        int  kb   = t2 * BK;
        bool full = (kb + BK <= K);
        if (full) { asyncA(kb, buf); asyncW(kb, buf); }
        else      { syncA(kb, buf);  syncW(kb, buf); }
    };

    v8f acc[2][4];
#pragma unroll
    for (int r = 0; r < 2; r++)
#pragma unroll
        for (int c = 0; c < 4; c++)
#pragma unroll
            for (int j = 0; j < 8; j++) acc[r][c][j] = 0.f;

    issueTile(0, 0);

    int lane = tid & 31, wid = tid >> 5;
    int wm = wid & 3, wn = wid >> 2;       // 4 M-waves x 2 N-waves
    int hf = lane >> 4, l15 = lane & 15;

    int kt = (K + BK - 1) / BK;
    for (int t = 0; t < kt; t++) {
        asm volatile("s_wait_asynccnt 0" ::: "memory");   // my async stores done
        __syncthreads();                                  // everyone's stores done
        if (t + 1 < kt) issueTile(t + 1, (t + 1) & 1);
        if (t + 2 < kt) {                                 // L2 prefetch 2 tiles out
            int pr = m0 + (tid >> 1); pr = pr < M - 1 ? pr : M - 1;
            __builtin_prefetch(Ag + (long)pr * lda + (t + 2) * BK, 0, 1);
        }
        int buf = t & 1;
#pragma unroll
        for (int kk = 0; kk < BK; kk += 32) {
            v16bf af[2], bfv[4];
#pragma unroll
            for (int r = 0; r < 2; r++) {
                int row = wm * 32 + r * 16 + l15;
                // A 16x32 bf16 fragment (ISA 7.12.2)
                bf16x8 lo = *(const bf16x8*)&AS(buf, row, kk + hf * 8);
                bf16x8 hi = *(const bf16x8*)&AS(buf, row, kk + 16 + hf * 8);
                af[r] = __builtin_shufflevector(lo, hi, 0,1,2,3,4,5,6,7,8,9,10,11,12,13,14,15);
            }
#pragma unroll
            for (int c = 0; c < 4; c++) {
                int col = wn * 64 + c * 16 + l15;
                // B 32x16 fragment: N = lane&15, lane<16 -> K 0..7,16..23, else K 8..15,24..31
                bf16x8 lo = *(const bf16x8*)&WS(buf, col, kk + hf * 8);
                bf16x8 hi = *(const bf16x8*)&WS(buf, col, kk + 16 + hf * 8);
                bfv[c] = __builtin_shufflevector(lo, hi, 0,1,2,3,4,5,6,7,8,9,10,11,12,13,14,15);
            }
#pragma unroll
            for (int r = 0; r < 2; r++)
#pragma unroll
                for (int c = 0; c < 4; c++)
                    acc[r][c] = __builtin_amdgcn_wmma_f32_16x16x32_bf16(
                        false, af[r], false, bfv[c], (short)0, acc[r][c], false, false);
        }
    }

    // C/D layout: VGPR j -> row j (lanes 0-15) / j+8 (lanes 16-31), col = lane&15
#pragma unroll
    for (int r = 0; r < 2; r++)
#pragma unroll
        for (int c = 0; c < 4; c++) {
            int col = n0 + wn * 64 + c * 16 + l15;
            if (col >= N) continue;
#pragma unroll
            for (int j = 0; j < 8; j++) {
                int row = m0 + wm * 32 + r * 16 + hf * 8 + j;
                if (row < M) {
                    float v = alpha * acc[r][c][j];
                    if constexpr (OBF)
                        ((__bf16*)Cv)[offC + (long)row * ldc + col] = (__bf16)v;
                    else
                        ((float*)Cv)[offC + (long)row * ldc + col] = v;
                }
            }
        }
#undef AS
#undef WS
}

// ------------------------------- host side ---------------------------------

extern "C" void kernel_launch(void* const* d_in, const int* in_sizes, int n_in,
                              void* d_out, int out_size, void* d_ws, size_t ws_size,
                              hipStream_t stream) {
    (void)in_sizes; (void)n_in; (void)out_size; (void)ws_size;
    constexpr int  Bc = C_B, SPc = C_SP, SSc = C_SS, Nc = C_N;
    constexpr int  DPc = C_DP, DSc = C_DS, DCc = C_DC;
    constexpr int  Hc = C_H, HDc = C_HD, HHc = C_HH, FPc = C_FP, FSc = C_FS;

    const float* prefix_x    = (const float*)d_in[0];
    const float* suffix_x    = (const float*)d_in[1];
    const float* cond        = (const float*)d_in[2];
    const float* p_ln_w      = (const float*)d_in[3];
    const float* p_post_ln_w = (const float*)d_in[8];
    const float* s_ada1_w    = (const float*)d_in[12];
    const float* s_ada1_b    = (const float*)d_in[13];
    const float* s_ada2_w    = (const float*)d_in[18];
    const float* s_ada2_b    = (const float*)d_in[19];

    char*  ws  = (char*)d_ws;
    size_t cur = 0;
    auto alloc = [&](size_t bytes) -> char* {
        char* p = ws + cur;
        cur = (cur + bytes + 255) & ~(size_t)255;
        return p;
    };
    dim3 blk(256);
    auto ewgrid = [](long n) { long g = (n + 255) / 256; return dim3((unsigned)(g > 8192 ? 8192 : g)); };
    auto toBf16 = [&](const void* src, long n) -> const __bf16* {
        __bf16* d = (__bf16*)alloc((size_t)n * 2);
        k_f32_to_bf16<<<ewgrid(n), blk, 0, stream>>>((const float*)src, d, n);
        return d;
    };
    auto ggrid = [](int M, int N, int z) {
        return dim3((unsigned)((N + BN - 1) / BN), (unsigned)((M + BM - 1) / BM), (unsigned)z);
    };
    const size_t SH = GEMM_LDS_BYTES;   // 73728 B dynamic LDS per block

    // ---- weights -> bf16 (once per launch) ----
    const __bf16* w_pq = toBf16(d_in[4],  (long)HHc * DPc);
    const __bf16* w_pk = toBf16(d_in[5],  (long)HDc * DPc);
    const __bf16* w_pv = toBf16(d_in[6],  (long)HDc * DPc);
    const __bf16* w_po = toBf16(d_in[7],  (long)DPc * HHc);
    const __bf16* w_pg = toBf16(d_in[9],  (long)FPc * DPc);
    const __bf16* w_pu = toBf16(d_in[10], (long)FPc * DPc);
    const __bf16* w_pd = toBf16(d_in[11], (long)DPc * FPc);
    const __bf16* w_sq = toBf16(d_in[14], (long)HHc * DSc);
    const __bf16* w_sk = toBf16(d_in[15], (long)HDc * DSc);
    const __bf16* w_sv = toBf16(d_in[16], (long)HDc * DSc);
    const __bf16* w_so = toBf16(d_in[17], (long)DSc * HHc);
    const __bf16* w_sg = toBf16(d_in[20], (long)FSc * DSc);
    const __bf16* w_su = toBf16(d_in[21], (long)FSc * DSc);
    const __bf16* w_sd = toBf16(d_in[22], (long)DSc * FSc);

    // ---- activations / scratch ----
    __bf16* hp    = (__bf16*)alloc((size_t)Bc * SPc * DPc * 2);
    __bf16* hs    = (__bf16*)alloc((size_t)Bc * SSc * DSc * 2);
    float*  mod1  = (float*)alloc((size_t)Bc * 3 * DSc * 4);
    float*  mod2  = (float*)alloc((size_t)Bc * 3 * DSc * 4);
    float*  gate1 = (float*)alloc((size_t)Bc * DSc * 4);
    float*  gate2 = (float*)alloc((size_t)Bc * DSc * 4);
    __bf16* qbuf  = (__bf16*)alloc((size_t)Bc * Nc * HHc * 2);
    __bf16* kbuf  = (__bf16*)alloc((size_t)Bc * Nc * HDc * 2);
    __bf16* vbuf  = (__bf16*)alloc((size_t)Bc * Nc * HDc * 2);
    __bf16* vt    = (__bf16*)alloc((size_t)Bc * HDc * Nc * 2);
    float*  scor  = (float*)alloc((size_t)Bc * Hc * Nc * Nc * 4);
    __bf16* attn  = (__bf16*)alloc((size_t)Bc * Hc * Nc * Nc * 2);
    __bf16* aout  = (__bf16*)alloc((size_t)Bc * Nc * HHc * 2);
    float*  tmp_p = (float*)alloc((size_t)Bc * SPc * DPc * 4);
    float*  res_p = (float*)alloc((size_t)Bc * SPc * DPc * 4);
    float*  tmp_s = (float*)alloc((size_t)Bc * SSc * DSc * 4);
    float*  res_s = (float*)alloc((size_t)Bc * SSc * DSc * 4);
    __bf16* hpost = (__bf16*)alloc((size_t)Bc * SPc * DPc * 2);
    __bf16* h2    = (__bf16*)alloc((size_t)Bc * SSc * DSc * 2);
    __bf16* gh    = (__bf16*)alloc((size_t)Bc * SPc * FPc * 2);
    __bf16* uh    = (__bf16*)alloc((size_t)Bc * SPc * FPc * 2);
    float*  mlp_p = (float*)alloc((size_t)Bc * SPc * DPc * 4);
    __bf16* sgh   = (__bf16*)alloc((size_t)Bc * SSc * FSc * 2);
    __bf16* suh   = (__bf16*)alloc((size_t)Bc * SSc * FSc * 2);
    float*  mlp_s = (float*)alloc((size_t)Bc * SSc * DSc * 4);

    const long MP = (long)Bc * SPc;   // 6400 prefix rows
    const long MS = (long)Bc * SSc;   // 384 suffix rows

    // ---- norms + modulation ----
    k_mod<<<dim3((Bc * 3 * DSc + 255) / 256), blk, 0, stream>>>(cond, s_ada1_w, s_ada1_b, mod1, Bc, 3 * DSc, DCc);
    k_gemma_norm<<<dim3((unsigned)MP), blk, 0, stream>>>(prefix_x, p_ln_w, hp, DPc);
    k_ada_norm<<<dim3((unsigned)MS), blk, 0, stream>>>(suffix_x, mod1, hs, gate1, SSc, DSc);

    // ---- QKV projections (batched over B; C has row offset for suffix) ----
    k_gemm<true><<<ggrid(SPc, HHc, Bc), blk, SH, stream>>>(
        hp, w_pq, (void*)qbuf, SPc, HHc, DPc, DPc, DPc, HHc, 1.f, 1,
        (long)SPc * DPc, 0, 0, 0, (long)Nc * HHc, 0);
    k_gemm<true><<<ggrid(SSc, HHc, Bc), blk, SH, stream>>>(
        hs, w_sq, (void*)(qbuf + (long)SPc * HHc), SSc, HHc, DSc, DSc, DSc, HHc, 1.f, 1,
        (long)SSc * DSc, 0, 0, 0, (long)Nc * HHc, 0);
    k_gemm<true><<<ggrid(SPc, HDc, Bc), blk, SH, stream>>>(
        hp, w_pk, (void*)kbuf, SPc, HDc, DPc, DPc, DPc, HDc, 1.f, 1,
        (long)SPc * DPc, 0, 0, 0, (long)Nc * HDc, 0);
    k_gemm<true><<<ggrid(SSc, HDc, Bc), blk, SH, stream>>>(
        hs, w_sk, (void*)(kbuf + (long)SPc * HDc), SSc, HDc, DSc, DSc, DSc, HDc, 1.f, 1,
        (long)SSc * DSc, 0, 0, 0, (long)Nc * HDc, 0);
    k_gemm<true><<<ggrid(SPc, HDc, Bc), blk, SH, stream>>>(
        hp, w_pv, (void*)vbuf, SPc, HDc, DPc, DPc, DPc, HDc, 1.f, 1,
        (long)SPc * DPc, 0, 0, 0, (long)Nc * HDc, 0);
    k_gemm<true><<<ggrid(SSc, HDc, Bc), blk, SH, stream>>>(
        hs, w_sv, (void*)(vbuf + (long)SPc * HDc), SSc, HDc, DSc, DSc, DSc, HDc, 1.f, 1,
        (long)SSc * DSc, 0, 0, 0, (long)Nc * HDc, 0);

    // ---- RoPE ----
    k_rope<<<ewgrid((long)Bc * Nc * Hc * (HDc / 2)), blk, 0, stream>>>(qbuf, Bc, Nc, Hc, HDc);
    k_rope<<<ewgrid((long)Bc * Nc * 1 * (HDc / 2)), blk, 0, stream>>>(kbuf, Bc, Nc, 1, HDc);

    // ---- V^T so attn@V uses the weight-form async fast path ----
    k_transpose_bf16<<<dim3((HDc + 31) / 32, (Nc + 31) / 32, Bc), blk, 0, stream>>>(
        vbuf, vt, Nc, HDc, (long)Nc * HDc, (long)HDc * Nc);

    // ---- attention scores: z = b*H + h ----
    k_gemm<false><<<ggrid(Nc, Nc, Bc * Hc), blk, SH, stream>>>(
        qbuf, kbuf, (void*)scor, Nc, Nc, HDc, HHc, HDc, Nc, 0.0625f, Hc,
        (long)Nc * HHc, HDc,
        (long)Nc * HDc, 0,
        (long)Hc * Nc * Nc, (long)Nc * Nc);
    k_softmax<<<dim3((unsigned)((long)Bc * Hc * Nc)), blk, 0, stream>>>(scor, attn, Nc, SPc);
    // ---- attn @ V  via V^T (N=HD rows, K=Nc cols) ----
    k_gemm<true><<<ggrid(Nc, HDc, Bc * Hc), blk, SH, stream>>>(
        attn, vt, (void*)aout, Nc, HDc, Nc, Nc, Nc, HHc, 1.f, Hc,
        (long)Hc * Nc * Nc, (long)Nc * Nc,
        (long)HDc * Nc, 0,
        (long)Nc * HHc, HDc);

    // ---- output projections + residuals ----
    k_gemm<false><<<ggrid(SPc, DPc, Bc), blk, SH, stream>>>(
        aout, w_po, (void*)tmp_p, SPc, DPc, HHc, HHc, HHc, DPc, 1.f, 1,
        (long)Nc * HHc, 0, 0, 0, (long)SPc * DPc, 0);
    k_add<<<ewgrid(MP * DPc), blk, 0, stream>>>(prefix_x, tmp_p, res_p, MP * DPc);
    k_gemm<false><<<ggrid(SSc, DSc, Bc), blk, SH, stream>>>(
        aout + (long)SPc * HHc, w_so, (void*)tmp_s, SSc, DSc, HHc, HHc, HHc, DSc, 1.f, 1,
        (long)Nc * HHc, 0, 0, 0, (long)SSc * DSc, 0);
    k_add_gate<<<ewgrid(MS * DSc), blk, 0, stream>>>(suffix_x, tmp_s, gate1, res_s, SSc, DSc, MS * DSc);

    // ---- second norms ----
    k_mod<<<dim3((Bc * 3 * DSc + 255) / 256), blk, 0, stream>>>(cond, s_ada2_w, s_ada2_b, mod2, Bc, 3 * DSc, DCc);
    k_gemma_norm<<<dim3((unsigned)MP), blk, 0, stream>>>(res_p, p_post_ln_w, hpost, DPc);
    k_ada_norm<<<dim3((unsigned)MS), blk, 0, stream>>>(res_s, mod2, h2, gate2, SSc, DSc);

    // ---- prefix MLP (the 1.3 TFLOP hot spot) ----
    k_gemm<true><<<ggrid((int)MP, FPc, 1), blk, SH, stream>>>(
        hpost, w_pg, (void*)gh, (int)MP, FPc, DPc, DPc, DPc, FPc, 1.f, 1, 0, 0, 0, 0, 0, 0);
    k_gemm<true><<<ggrid((int)MP, FPc, 1), blk, SH, stream>>>(
        hpost, w_pu, (void*)uh, (int)MP, FPc, DPc, DPc, DPc, FPc, 1.f, 1, 0, 0, 0, 0, 0, 0);
    k_gelumul<<<ewgrid(MP * FPc), blk, 0, stream>>>(gh, uh, gh, MP * FPc);
    k_gemm<false><<<ggrid((int)MP, DPc, 1), blk, SH, stream>>>(
        gh, w_pd, (void*)mlp_p, (int)MP, DPc, FPc, FPc, FPc, DPc, 1.f, 1, 0, 0, 0, 0, 0, 0);
    k_add<<<ewgrid(MP * DPc), blk, 0, stream>>>(res_p, mlp_p, (float*)d_out, MP * DPc);

    // ---- suffix MLP ----
    k_gemm<true><<<ggrid((int)MS, FSc, 1), blk, SH, stream>>>(
        h2, w_sg, (void*)sgh, (int)MS, FSc, DSc, DSc, DSc, FSc, 1.f, 1, 0, 0, 0, 0, 0, 0);
    k_gemm<true><<<ggrid((int)MS, FSc, 1), blk, SH, stream>>>(
        h2, w_su, (void*)suh, (int)MS, FSc, DSc, DSc, DSc, FSc, 1.f, 1, 0, 0, 0, 0, 0, 0);
    k_gelumul<<<ewgrid(MS * FSc), blk, 0, stream>>>(sgh, suh, sgh, MS * FSc);
    k_gemm<false><<<ggrid((int)MS, DSc, 1), blk, SH, stream>>>(
        sgh, w_sd, (void*)mlp_s, (int)MS, DSc, FSc, FSc, FSc, DSc, 1.f, 1, 0, 0, 0, 0, 0, 0);
    k_add_gate<<<ewgrid(MS * DSc), blk, 0, stream>>>(
        res_s, mlp_s, gate2, (float*)d_out + MP * DPc, SSc, DSc, MS * DSc);
}
